// Set2Set_90134183674518
// MI455X (gfx1250) — matrix operand order repaired and checked by reading, not compiled, not run on hardware
//
#include <hip/hip_runtime.h>
#include <hip/hip_bf16.h>

// ---------------------------------------------------------------------------
// Set2Set (6 iters, 3-layer LSTM on 10000 segments + segment softmax attention
// over 500000x256 feats) for gfx1250 (MI455X).
// GEMMs use v_wmma_f32_16x16x32_f16 with the shared weight chunk staged in LDS
// (5-wave workgroups, double-staged through registers so global loads overlap
// WMMA). Attention uses coalesced float4 loads and wave32 shuffle reductions;
// sorted segment_ids -> per-segment waves, 0 atomics.
// ---------------------------------------------------------------------------

#define IN_DIM    256
#define OUT_DIM   512          // 2*IN_DIM
#define GATES_DIM 1024         // 4*IN_DIM
#define NUM_ITERS 6
#define NUM_LAYERS 3
#define NUM_NODES 500000
#define NUM_SEG   10000

#define GEMM_WAVES 5           // waves per workgroup (625 M-tiles = 125 * 5)
#define BSTRIDE    40          // padded halfs per B column in LDS (bank-safe)

typedef _Float16 v8h  __attribute__((ext_vector_type(8)));
typedef _Float16 v16h __attribute__((ext_vector_type(16)));
typedef float    v8f  __attribute__((ext_vector_type(8)));

// ---------------- weight / bias packing (once per launch) -------------------

// W16[n*K + k] = (k < in_sz) ? w_ih[n][k] : w_hh[n][k-in_sz], converted to f16.
__global__ void wpack_kernel(const float* __restrict__ wih,
                             const float* __restrict__ whh,
                             _Float16* __restrict__ W16,
                             int in_sz, int K) {
  int k = blockIdx.x * blockDim.x + threadIdx.x;
  int n = blockIdx.y;
  float v = (k < in_sz) ? wih[(size_t)n * in_sz + k]
                        : whh[(size_t)n * IN_DIM + (k - in_sz)];
  W16[(size_t)n * K + k] = (_Float16)v;
}

__global__ void bias_kernel(const float* __restrict__ bi,
                            const float* __restrict__ bh,
                            float* __restrict__ out) {
  int n = blockIdx.x * blockDim.x + threadIdx.x;
  if (n < GATES_DIM) out[n] = bi[n] + bh[n];
}

// ---------------- segment offsets (segment_ids is sorted) -------------------

__global__ void seg_starts_kernel(const int* __restrict__ seg,
                                  int* __restrict__ start, int n, int nseg) {
  int i = blockIdx.x * blockDim.x + threadIdx.x;
  if (i >= n) return;
  int cur  = seg[i];
  int prev = (i == 0) ? -1 : seg[i - 1];
  for (int s = prev + 1; s <= cur; ++s) start[s] = i;
  if (i == n - 1)
    for (int s = cur + 1; s <= nseg; ++s) start[s] = n;
}

// ---------------- activation pack: A16 = f16([x | h_l]) ---------------------

__global__ void packA_kernel(const float* __restrict__ x, int xstride,
                             const float* __restrict__ hl,
                             _Float16* __restrict__ A16,
                             int in_sz, int K) {
  int k = blockIdx.x * blockDim.x + threadIdx.x;
  int m = blockIdx.y;
  float v = (k < in_sz) ? x[(size_t)m * xstride + k]
                        : hl[(size_t)m * IN_DIM + (k - in_sz)];
  A16[(size_t)m * K + k] = (_Float16)v;
}

// ---------------- WMMA GEMM: gates(10000x1024) = A16 @ W16^T + bias ---------
// Workgroup = 5 waves -> 80x64 output strip; grid = (125, 16).
// Per K-step: the 64x32 f16 weight chunk (shared by all 5 waves) is staged
// into LDS; next chunk's global loads are issued before the WMMAs so L2
// latency overlaps compute. Fragment addressing per CDNA5 ISA 7.12.2 (wave32).

template <int K>
__global__ __launch_bounds__(32 * GEMM_WAVES)
void gemm_wmma_kernel(const _Float16* __restrict__ A,
                      const _Float16* __restrict__ W,
                      const float* __restrict__ bias,
                      float* __restrict__ out) {
  __shared__ _Float16 Bs[64 * BSTRIDE];

  const int tid    = threadIdx.x;
  const int lane   = tid & 31;
  const int wave   = tid >> 5;                // 0..4
  const int l15    = lane & 15;
  const int half   = (lane >> 4) & 1;
  const int mt     = blockIdx.x * GEMM_WAVES + wave;  // 0..624
  const int nbase  = blockIdx.y * 64;
  const int row    = mt * 16 + l15;           // A row for this lane
  const int aShift = half * 8;                // A-frag K shift (halfs)
  const int bShift = half * 8;                // B-frag K shift (halfs)

  v8f acc[4];
#pragma unroll
  for (int t = 0; t < 4; ++t) {
    float b = bias[nbase + t * 16 + l15];
    acc[t] = (v8f){b, b, b, b, b, b, b, b};
  }

  const _Float16* Arow = A + (size_t)row * K;

  // B staging: 64 cols x 32 halfs = 256 16-byte chunks, 160 threads.
  const int q0 = tid;           // always valid (q0 < 256)
  const int q1 = tid + 160;     // valid only for tid < 96
  const int c0 = q0 >> 2, p0 = q0 & 3;
  const int c1 = q1 >> 2, p1 = q1 & 3;

  uint4 r0 = *(const uint4*)(W + (size_t)(nbase + c0) * K + p0 * 8);
  uint4 r1 = {0, 0, 0, 0};
  if (q1 < 256) r1 = *(const uint4*)(W + (size_t)(nbase + c1) * K + p1 * 8);

  for (int kk = 0; kk < K; kk += 32) {
    // commit staged chunk to LDS
    *(uint4*)(Bs + c0 * BSTRIDE + p0 * 8) = r0;
    if (q1 < 256) *(uint4*)(Bs + c1 * BSTRIDE + p1 * 8) = r1;
    __syncthreads();

    // issue next chunk's global loads before computing (overlaps WMMAs)
    if (kk + 32 < K) {
      r0 = *(const uint4*)(W + (size_t)(nbase + c0) * K + kk + 32 + p0 * 8);
      if (q1 < 256)
        r1 = *(const uint4*)(W + (size_t)(nbase + c1) * K + kk + 32 + p1 * 8);
      __builtin_prefetch(Arow + kk + 64, 0, 3);
    }

    // A 16x32 f16 fragment: VGPR0..3 = K[kk+aShift .. +8), VGPR4..7 = +16
    v8h alo = *(const v8h*)(Arow + kk + aShift);
    v8h ahi = *(const v8h*)(Arow + kk + 16 + aShift);
    v16h a = __builtin_shufflevector(alo, ahi, 0, 1, 2, 3, 4, 5, 6, 7,
                                     8, 9, 10, 11, 12, 13, 14, 15);
#pragma unroll
    for (int t = 0; t < 4; ++t) {
      // B 32x16 fragment from LDS: lane holds column t*16+l15, 16 K-halfs
      const _Float16* bp = Bs + (t * 16 + l15) * BSTRIDE + bShift;
      v8h blo = *(const v8h*)bp;
      v8h bhi = *(const v8h*)(bp + 16);
      v16h b = __builtin_shufflevector(blo, bhi, 0, 1, 2, 3, 4, 5, 6, 7,
                                       8, 9, 10, 11, 12, 13, 14, 15);
      acc[t] = __builtin_amdgcn_wmma_f32_16x16x32_f16(
          /*neg_a=*/false, a, /*neg_b=*/false, b,
          /*c_mod=*/(short)0, acc[t], /*reuse_a=*/false, /*reuse_b=*/false);
    }
    __syncthreads();
  }

  // D 16x16 f32 layout: VGPR r -> M = half*8 + r, N = l15
  const int mBase = mt * 16 + half * 8;
#pragma unroll
  for (int t = 0; t < 4; ++t) {
    int n = nbase + t * 16 + l15;
#pragma unroll
    for (int r = 0; r < 8; ++r)
      out[(size_t)(mBase + r) * GATES_DIM + n] = acc[t][r];
  }
}

// ---------------- LSTM gate activation --------------------------------------

__device__ __forceinline__ float sigmoidf_(float x) {
  return 1.0f / (1.0f + __expf(-x));
}

__global__ void gate_kernel(const float* __restrict__ gates,
                            float* __restrict__ cbuf,
                            float* __restrict__ hbuf,
                            float* __restrict__ qstar /* null unless layer 2 */) {
  int idx = blockIdx.x * blockDim.x + threadIdx.x;   // [0, 10000*256)
  int m = idx >> 8, d = idx & 255;
  const float* g = gates + (size_t)m * GATES_DIM;
  float i_ = sigmoidf_(g[d]);
  float f_ = sigmoidf_(g[IN_DIM + d]);
  float g_ = tanhf(g[2 * IN_DIM + d]);
  float o_ = sigmoidf_(g[3 * IN_DIM + d]);
  float cn = f_ * cbuf[idx] + i_ * g_;
  float hn = o_ * tanhf(cn);
  cbuf[idx] = cn;
  hbuf[idx] = hn;
  if (qstar) qstar[(size_t)m * OUT_DIM + d] = hn;
}

// ---------------- attention: e[i] = dot(feats[i], q[seg[i]]) ----------------
// One wave per node; each lane covers 8 dims (two float4 loads, coalesced).

__global__ void e_kernel(const float* __restrict__ feats,
                         const int* __restrict__ seg,
                         const float* __restrict__ qstar,
                         float* __restrict__ ebuf) {
  int g = blockIdx.x * blockDim.x + threadIdx.x;
  int node = g >> 5, lane = g & 31;
  if (node >= NUM_NODES) return;
  const float4* fr = (const float4*)(feats + (size_t)node * IN_DIM) + lane * 2;
  int s = seg[node];
  const float4* qr = (const float4*)(qstar + (size_t)s * OUT_DIM) + lane * 2;
  float4 f0 = fr[0], f1 = fr[1];
  float4 q0 = qr[0], q1 = qr[1];
  float d = f0.x * q0.x + f0.y * q0.y + f0.z * q0.z + f0.w * q0.w +
            f1.x * q1.x + f1.y * q1.y + f1.z * q1.z + f1.w * q1.w;
#pragma unroll
  for (int off = 16; off > 0; off >>= 1) d += __shfl_xor(d, off, 32);
  if (lane == 0) ebuf[node] = d;
}

// ---------------- segment max + sum(exp) (wave per segment) -----------------

__global__ void mden_kernel(const float* __restrict__ ebuf,
                            const int* __restrict__ start,
                            float* __restrict__ mbuf,
                            float* __restrict__ dbuf) {
  int g = blockIdx.x * blockDim.x + threadIdx.x;
  int s = g >> 5, lane = g & 31;
  if (s >= NUM_SEG) return;
  int b0 = start[s], e0 = start[s + 1];
  float mx = -INFINITY;
  for (int j = b0 + lane; j < e0; j += 32) mx = fmaxf(mx, ebuf[j]);
#pragma unroll
  for (int off = 16; off > 0; off >>= 1) mx = fmaxf(mx, __shfl_xor(mx, off, 32));
  float sm = 0.0f;
  for (int j = b0 + lane; j < e0; j += 32) sm += __expf(ebuf[j] - mx);
#pragma unroll
  for (int off = 16; off > 0; off >>= 1) sm += __shfl_xor(sm, off, 32);
  if (lane == 0) { mbuf[s] = mx; dbuf[s] = sm; }
}

// ---------------- r[s] = sum alpha*feats -> q_star[:,256:512] ---------------
// Wave per segment; lane accumulates 8 dims in registers, one store at end.

__global__ void r_kernel(const float* __restrict__ feats,
                         const float* __restrict__ ebuf,
                         const int* __restrict__ start,
                         const float* __restrict__ mbuf,
                         const float* __restrict__ dbuf,
                         float* __restrict__ qstar) {
  int g = blockIdx.x * blockDim.x + threadIdx.x;
  int s = g >> 5, lane = g & 31;
  if (s >= NUM_SEG) return;
  int b0 = start[s], e0 = start[s + 1];
  float mx = mbuf[s];
  float rd = 1.0f / dbuf[s];          // unused when segment empty
  float4 a0 = {0, 0, 0, 0}, a1 = {0, 0, 0, 0};
  for (int j = b0; j < e0; ++j) {
    float w = __expf(ebuf[j] - mx) * rd;
    const float4* fr = (const float4*)(feats + (size_t)j * IN_DIM) + lane * 2;
    float4 f0 = fr[0], f1 = fr[1];
    a0.x += w * f0.x; a0.y += w * f0.y; a0.z += w * f0.z; a0.w += w * f0.w;
    a1.x += w * f1.x; a1.y += w * f1.y; a1.z += w * f1.z; a1.w += w * f1.w;
  }
  float4* dst = (float4*)(qstar + (size_t)s * OUT_DIM + IN_DIM) + lane * 2;
  dst[0] = a0;
  dst[1] = a1;
}

// ---------------------------------------------------------------------------

extern "C" void kernel_launch(void* const* d_in, const int* in_sizes, int n_in,
                              void* d_out, int out_size, void* d_ws, size_t ws_size,
                              hipStream_t stream) {
  const float* feats = (const float*)d_in[0];
  const int*   seg   = (const int*)d_in[1];
  const float* wih[NUM_LAYERS] = {(const float*)d_in[2], (const float*)d_in[6],
                                  (const float*)d_in[10]};
  const float* whh[NUM_LAYERS] = {(const float*)d_in[3], (const float*)d_in[7],
                                  (const float*)d_in[11]};
  const float* bih[NUM_LAYERS] = {(const float*)d_in[4], (const float*)d_in[8],
                                  (const float*)d_in[12]};
  const float* bhh[NUM_LAYERS] = {(const float*)d_in[5], (const float*)d_in[9],
                                  (const float*)d_in[13]};
  float* qstar = (float*)d_out;                 // (10000, 512) == final output

  const int Kl[NUM_LAYERS]   = {OUT_DIM + IN_DIM, 2 * IN_DIM, 2 * IN_DIM}; // 768,512,512
  const int insz[NUM_LAYERS] = {OUT_DIM, IN_DIM, IN_DIM};                  // 512,256,256

  // ---- workspace carve-up (all 256B aligned) ----
  char* ws = (char*)d_ws;
  auto carve = [&](size_t bytes) {
    void* p = (void*)ws;
    ws += (bytes + 255) & ~(size_t)255;
    return p;
  };
  _Float16* W16[NUM_LAYERS];
  float*    biasS[NUM_LAYERS];
  for (int l = 0; l < NUM_LAYERS; ++l)
    W16[l] = (_Float16*)carve((size_t)GATES_DIM * Kl[l] * sizeof(_Float16));
  for (int l = 0; l < NUM_LAYERS; ++l)
    biasS[l] = (float*)carve(GATES_DIM * sizeof(float));
  float* hbuf = (float*)carve((size_t)NUM_LAYERS * NUM_SEG * IN_DIM * sizeof(float));
  float* cbuf = (float*)carve((size_t)NUM_LAYERS * NUM_SEG * IN_DIM * sizeof(float));
  _Float16* A16 = (_Float16*)carve((size_t)NUM_SEG * Kl[0] * sizeof(_Float16));
  float* gates = (float*)carve((size_t)NUM_SEG * GATES_DIM * sizeof(float));
  float* ebuf  = (float*)carve((size_t)NUM_NODES * sizeof(float));
  float* mbuf  = (float*)carve(NUM_SEG * sizeof(float));
  float* dbuf  = (float*)carve(NUM_SEG * sizeof(float));
  int*   segst = (int*)carve((NUM_SEG + 1) * sizeof(int));
  (void)ws_size; (void)in_sizes; (void)n_in; (void)out_size;

  // ---- once-per-launch prep ----
  hipMemsetAsync(hbuf, 0, (size_t)NUM_LAYERS * NUM_SEG * IN_DIM * sizeof(float), stream);
  hipMemsetAsync(cbuf, 0, (size_t)NUM_LAYERS * NUM_SEG * IN_DIM * sizeof(float), stream);
  hipMemsetAsync(qstar, 0, (size_t)NUM_SEG * OUT_DIM * sizeof(float), stream);

  for (int l = 0; l < NUM_LAYERS; ++l) {
    wpack_kernel<<<dim3(Kl[l] / 256, GATES_DIM), 256, 0, stream>>>(
        wih[l], whh[l], W16[l], insz[l], Kl[l]);
    bias_kernel<<<GATES_DIM / 256, 256, 0, stream>>>(bih[l], bhh[l], biasS[l]);
  }
  seg_starts_kernel<<<(NUM_NODES + 255) / 256, 256, 0, stream>>>(
      seg, segst, NUM_NODES, NUM_SEG);

  // ---- 6 set2set iterations ----
  for (int it = 0; it < NUM_ITERS; ++it) {
    for (int l = 0; l < NUM_LAYERS; ++l) {
      const float* x  = (l == 0) ? qstar : (hbuf + (size_t)(l - 1) * NUM_SEG * IN_DIM);
      int xstride     = (l == 0) ? OUT_DIM : IN_DIM;
      float* hl       = hbuf + (size_t)l * NUM_SEG * IN_DIM;
      float* cl       = cbuf + (size_t)l * NUM_SEG * IN_DIM;

      packA_kernel<<<dim3(Kl[l] / 256, NUM_SEG), 256, 0, stream>>>(
          x, xstride, hl, A16, insz[l], Kl[l]);

      if (l == 0)
        gemm_wmma_kernel<768>
            <<<dim3(NUM_SEG / 16 / GEMM_WAVES, GATES_DIM / 64),
               32 * GEMM_WAVES, 0, stream>>>(A16, W16[l], biasS[l], gates);
      else
        gemm_wmma_kernel<512>
            <<<dim3(NUM_SEG / 16 / GEMM_WAVES, GATES_DIM / 64),
               32 * GEMM_WAVES, 0, stream>>>(A16, W16[l], biasS[l], gates);

      gate_kernel<<<(NUM_SEG * IN_DIM) / 256, 256, 0, stream>>>(
          gates, cl, hl, (l == NUM_LAYERS - 1) ? qstar : nullptr);
    }

    e_kernel<<<(NUM_NODES * 32) / 256, 256, 0, stream>>>(feats, seg, qstar, ebuf);
    mden_kernel<<<(NUM_SEG * 32) / 256, 256, 0, stream>>>(ebuf, segst, mbuf, dbuf);
    r_kernel<<<(NUM_SEG * 32) / 256, 256, 0, stream>>>(
        feats, ebuf, segst, mbuf, dbuf, qstar);
  }
}
